// BertDiora_24721831756130
// MI455X (gfx1250) — compile-verified, matrix operand order checked by reading, not run on hardware
//
#include <hip/hip_runtime.h>

#define TT   32
#define BB   64
#define DD   768
#define D2   1536
#define TRI  528
#define ROWC 1538   // 768 inside vec + 768 outside vec + inside score + outside score

typedef __attribute__((ext_vector_type(16))) __bf16 v16bf;
typedef __attribute__((ext_vector_type(8)))  float  v8f;

union Frag { v16bf v; unsigned short s[16]; uint4 q[2]; };

__device__ __forceinline__ unsigned short f2bf(float x) {
    unsigned int u = __float_as_uint(x);
    unsigned int r = u + 0x7FFFu + ((u >> 16) & 1u);   // round-to-nearest-even
    return (unsigned short)(r >> 16);
}

__device__ __forceinline__ int cellIdx(int b0, int L) {
    // flatten_tri(b0, b0+L, 32)
    return TRI - ((34 - L) * (33 - L)) / 2 + b0;
}
__device__ __forceinline__ size_t cellOff(int idx, int b) {
    return ((size_t)(idx * BB + b)) * ROWC;
}

// left/parent cell (li) and right/sister cell (ri) for span s, context c
__device__ __forceinline__ void childCells(int pass, int len, int c, int s, int& li, int& ri) {
    if (pass == 0) {                       // inside: children (s,s+j) and (s+j,s+len)
        int j = c + 1;
        li = cellIdx(s, j);
        ri = cellIdx(s + j, len - j);
    } else {                               // outside: parent + sister
        int C = TT - len; int j = c + 1; int k = C - c;
        if (s > c) { li = cellIdx(s - j, len + j); ri = cellIdx(s - j, j);   }
        else       { li = cellIdx(s, len + k);     ri = cellIdx(s + len, k); }
    }
}

__device__ __forceinline__ float waveRed(float v) {
    #pragma unroll
    for (int m = 16; m > 0; m >>= 1) v += __shfl_xor(v, m, 32);
    return v;
}

// ---------------------------------------------------------------------------
// WMMA GEMM: out[r,n] = epilogue( sum_k A[r,k] * W[n,k] )
// AMODE: 0 = chart gather [left|right] (K=1536)
//        1 = chart gather right-only   (K=768)
//        2 = contiguous bf16 A (h)
//        3 = contiguous fp32 A (embeddings)
// EPI:   0 = store f32           (outF)
//        1 = tanh(x+bias) f32    (outF)
//        2 = relu(x+bias) bf16   (outH)
//        3 = outF += w * relu(x+bias)   (w from wts[(s*C+c)*64+b])
// block = 256 threads = 8 waves: 4 waves along M (64 rows), 2 along N (128 cols)
// each wave: 16x64 tile = 4 accumulators; grid = (rows/64, 768/128)
// ---------------------------------------------------------------------------
template<int AMODE, int EPI>
__global__ __launch_bounds__(256) void gemm_k(
    const float* __restrict__ chart, const unsigned short* __restrict__ Abf,
    const unsigned short* __restrict__ Wbf, const float* __restrict__ bias,
    float* __restrict__ outF, unsigned short* __restrict__ outH,
    const float* __restrict__ wts, int K, int len, int c, int pass, int C)
{
    const int lane  = threadIdx.x & 31;
    const int wid   = threadIdx.x >> 5;
    const int rBase = blockIdx.x * 64  + (wid & 3) * 16;
    const int nBase = blockIdx.y * 128 + (wid >> 2) * 64;
    const int koff  = (lane & 16) ? 8 : 0;     // A/B 16-bit fragment K-group split

    // ---- per-lane A row pointers ----
    const int mr = rBase + (lane & 15);
    const float* lp = nullptr; const float* rp = nullptr;
    const float* ap = nullptr; const unsigned short* ap16 = nullptr;
    if (AMODE == 0 || AMODE == 1) {
        int s = mr >> 6, b = mr & 63, li, ri;
        childCells(pass, len, c, s, li, ri);
        lp = chart + cellOff(li, b) + (pass ? DD : 0);  // parent uses outside vec
        rp = chart + cellOff(ri, b);                    // right/sister: inside vec
    } else if (AMODE == 2) {
        ap16 = Abf + (size_t)mr * K;
    } else {
        ap = chart + (size_t)mr * K;                    // chart param doubles as fp32 A
    }

    // ---- B row pointers (weights, row-major [N,K] bf16) ----
    const unsigned short* br[4];
    #pragma unroll
    for (int j = 0; j < 4; ++j)
        br[j] = Wbf + (size_t)(nBase + j * 16 + (lane & 15)) * K;

    v8f acc[4] = {};
    for (int kb = 0; kb < K; kb += 32) {
        Frag a;
        if (AMODE == 2) {
            a.q[0] = *(const uint4*)(ap16 + kb + koff);
            a.q[1] = *(const uint4*)(ap16 + kb + koff + 16);
        } else {
            const float* src; int kk;
            if (AMODE == 0) { if (kb < DD) { src = lp; kk = kb; } else { src = rp; kk = kb - DD; } }
            else if (AMODE == 1) { src = rp; kk = kb; }
            else                 { src = ap; kk = kb; }
            #pragma unroll
            for (int h = 0; h < 2; ++h) {
                const float* p = src + kk + koff + h * 16;
                #pragma unroll
                for (int i = 0; i < 4; ++i) {
                    float2 f = *(const float2*)(p + 2 * i);  // chart rows are 8B aligned
                    a.s[h * 8 + 2 * i]     = f2bf(f.x);
                    a.s[h * 8 + 2 * i + 1] = f2bf(f.y);
                }
            }
        }
        #pragma unroll
        for (int j = 0; j < 4; ++j) {
            Frag bf;
            bf.q[0] = *(const uint4*)(br[j] + kb + koff);
            bf.q[1] = *(const uint4*)(br[j] + kb + koff + 16);
            acc[j] = __builtin_amdgcn_wmma_f32_16x16x32_bf16(
                         false, a.v, false, bf.v, (short)0, acc[j], false, false);
        }
    }

    // ---- epilogue: lane owns col n = nBase+j*16+(lane&15), rows v + 8*(lane>=16) ----
    const int nLane = lane & 15;
    const int rHi   = (lane >> 4) * 8;
    #pragma unroll
    for (int j = 0; j < 4; ++j) {
        int n = nBase + j * 16 + nLane;
        float bn = (EPI != 0) ? bias[n] : 0.f;
        #pragma unroll
        for (int v = 0; v < 8; ++v) {
            int r = rBase + rHi + v;
            float val = acc[j][v];
            if (EPI == 0) {
                outF[(size_t)r * DD + n] = val;
            } else if (EPI == 1) {
                outF[(size_t)r * DD + n] = tanhf(val + bn);
            } else if (EPI == 2) {
                float t = val + bn; t = t > 0.f ? t : 0.f;
                outH[(size_t)r * DD + n] = f2bf(t);
            } else {
                float t = val + bn; t = t > 0.f ? t : 0.f;
                int s = r >> 6, b = r & 63;
                float wgt = wts[((size_t)s * C + c) * BB + b];
                outF[(size_t)r * DD + n] += wgt * t;
            }
        }
    }
}

// score[s,c,b] = dot(left, u_row) + child scores   (one wave per row)
__global__ __launch_bounds__(256) void score_dot_k(
    const float* __restrict__ chart, const float* __restrict__ u,
    float* __restrict__ scores, int len, int c, int pass, int C)
{
    const int lane = threadIdx.x & 31;
    const int r    = (blockIdx.x * 256 + threadIdx.x) >> 5;
    const int s = r >> 6, b = r & 63;
    int li, ri; childCells(pass, len, c, s, li, ri);
    const float* lp   = chart + cellOff(li, b) + (pass ? DD : 0);
    const float* urow = u + (size_t)r * DD;
    float sum = 0.f;
    #pragma unroll 4
    for (int t = 0; t < DD / 32; ++t) { int i = lane + t * 32; sum += lp[i] * urow[i]; }
    sum = waveRed(sum);
    if (lane == 0) {
        float sl = chart[cellOff(li, b) + (pass ? 1537 : 1536)];
        float sr = chart[cellOff(ri, b) + 1536];
        scores[((size_t)s * C + c) * BB + b] = sum + sl + sr;
    }
}

// softmax over contexts; weights stored back in-place; writes chart score
__global__ __launch_bounds__(256) void softmax_k(
    float* __restrict__ chart, float* __restrict__ scores,
    int len, int pass, int S, int C)
{
    int t = blockIdx.x * 256 + threadIdx.x;
    if (t >= S * BB) return;
    int s = t >> 6, b = t & 63;
    float sv[32];
    float mx = -3.4e38f;
    for (int c = 0; c < C; ++c) {
        sv[c] = scores[((size_t)s * C + c) * BB + b];
        mx = fmaxf(mx, sv[c]);
    }
    float se = 0.f;
    for (int c = 0; c < C; ++c) se += __expf(sv[c] - mx);
    float inv = 1.f / se;
    float outsc = 0.f;
    for (int c = 0; c < C; ++c) {
        float w = __expf(sv[c] - mx) * inv;
        outsc += w * sv[c];
        scores[((size_t)s * C + c) * BB + b] = w;
    }
    int ci = cellIdx(s, len);
    chart[cellOff(ci, b) + (pass ? 1537 : 1536)] = outsc;
}

// L2-normalize accumulated vectors into the chart (one wave per row)
__global__ __launch_bounds__(256) void norm_k(
    float* __restrict__ chart, const float* __restrict__ src, int len, int pass)
{
    const int lane = threadIdx.x & 31;
    const int r    = (blockIdx.x * 256 + threadIdx.x) >> 5;
    const int s = r >> 6, b = r & 63;
    const float* arow = src + (size_t)r * DD;
    float ss = 0.f;
    #pragma unroll 4
    for (int t = 0; t < DD / 32; ++t) { float x = arow[lane + t * 32]; ss += x * x; }
    ss = waveRed(ss);
    float denom = fmaxf(sqrtf(ss), 1e-12f);
    int ci = cellIdx(s, len);
    float* dst = chart + cellOff(ci, b) + (pass ? DD : 0);
    #pragma unroll 4
    for (int t = 0; t < DD / 32; ++t) { int i = lane + t * 32; dst[i] = arow[i] / denom; }
}

__global__ void zero_k(float* __restrict__ p, long long n) {
    long long i  = (long long)blockIdx.x * blockDim.x + threadIdx.x;
    long long st = (long long)gridDim.x * blockDim.x;
    for (; i < n; i += st) p[i] = 0.f;
}

__global__ void cvt_k(const float* __restrict__ s, unsigned short* __restrict__ d, int n) {
    int i = blockIdx.x * blockDim.x + threadIdx.x;
    if (i < n) d[i] = f2bf(s[i]);
}

// root: outside_vecs[527,b,:] = root_bias / ||root_bias||   (single block)
__global__ void root_init_k(float* __restrict__ chart, const float* __restrict__ rb) {
    __shared__ float red[256];
    int tid = threadIdx.x;
    float ps = 0.f;
    for (int i = tid; i < DD; i += 256) ps += rb[i] * rb[i];
    red[tid] = ps; __syncthreads();
    for (int s2 = 128; s2 > 0; s2 >>= 1) { if (tid < s2) red[tid] += red[tid + s2]; __syncthreads(); }
    float denom = fmaxf(sqrtf(red[0]), 1e-12f);
    for (int e = tid; e < BB * DD; e += 256) {
        int b = e / DD, i = e % DD;
        chart[cellOff(TRI - 1, b) + DD + i] = rb[i] / denom;
    }
}

extern "C" void kernel_launch(void* const* d_in, const int* in_sizes, int n_in,
                              void* d_out, int out_size, void* d_ws, size_t ws_size,
                              hipStream_t stream)
{
    const float* emb   = (const float*)d_in[0];
    const float* Wl    = (const float*)d_in[1];
    const float* bl    = (const float*)d_in[2];
    const float* W1    = (const float*)d_in[3];
    const float* b1    = (const float*)d_in[4];
    const float* W2    = (const float*)d_in[5];
    const float* b2    = (const float*)d_in[6];
    const float* Wb    = (const float*)d_in[7];
    const float* rootb = (const float*)d_in[8];
    float* chart = (float*)d_out;   // charts live directly in the output tensor

    char* w = (char*)d_ws;
    unsigned short* Wl_bf = (unsigned short*)w; w += (size_t)DD * DD * 2;
    unsigned short* Wb_bf = (unsigned short*)w; w += (size_t)DD * DD * 2;
    unsigned short* W1_bf = (unsigned short*)w; w += (size_t)DD * D2 * 2;
    unsigned short* W2_bf = (unsigned short*)w; w += (size_t)DD * DD * 2;
    float*          u     = (float*)w;          w += (size_t)2048 * DD * 4;
    unsigned short* h     = (unsigned short*)w; w += (size_t)2048 * DD * 2;
    float*          accv  = (float*)w;          w += (size_t)2048 * DD * 4;
    float*          scrs  = (float*)w;          w += (size_t)992 * BB * 4;
    (void)ws_size; (void)in_sizes; (void)n_in; (void)out_size;

    zero_k<<<4096, 256, 0, stream>>>(chart, (long long)TRI * BB * ROWC);
    cvt_k<<<(DD * DD + 255) / 256, 256, 0, stream>>>(Wl, Wl_bf, DD * DD);
    cvt_k<<<(DD * DD + 255) / 256, 256, 0, stream>>>(Wb, Wb_bf, DD * DD);
    cvt_k<<<(DD * D2 + 255) / 256, 256, 0, stream>>>(W1, W1_bf, DD * D2);
    cvt_k<<<(DD * DD + 255) / 256, 256, 0, stream>>>(W2, W2_bf, DD * DD);

    // leaves: normalize(tanh(emb @ Wl^T + bl)) -> inside chart cells 0..31
    gemm_k<3, 1><<<dim3(32, 6), 256, 0, stream>>>(emb, nullptr, Wl_bf, bl, u,
                                                  nullptr, nullptr, DD, 0, 0, 0, 0);
    norm_k<<<2048 / 8, 256, 0, stream>>>(chart, u, 1, 0);

    // ---------------- inside pass ----------------
    for (int len = 2; len <= TT; ++len) {
        int S = TT + 1 - len, C = len - 1, rows = S * BB;
        for (int c = 0; c < C; ++c) {
            gemm_k<1, 0><<<dim3(S, 6), 256, 0, stream>>>(chart, nullptr, Wb_bf, nullptr,
                                                         u, nullptr, nullptr, DD, len, c, 0, C);
            score_dot_k<<<rows / 8, 256, 0, stream>>>(chart, u, scrs, len, c, 0, C);
        }
        softmax_k<<<(rows + 255) / 256, 256, 0, stream>>>(chart, scrs, len, 0, S, C);
        zero_k<<<1024, 256, 0, stream>>>(accv, (long long)rows * DD);
        for (int c = 0; c < C; ++c) {
            gemm_k<0, 2><<<dim3(S, 6), 256, 0, stream>>>(chart, nullptr, W1_bf, b1,
                                                         nullptr, h, nullptr, D2, len, c, 0, C);
            gemm_k<2, 3><<<dim3(S, 6), 256, 0, stream>>>(chart, h, W2_bf, b2,
                                                         accv, nullptr, scrs, DD, len, c, 0, C);
        }
        norm_k<<<rows / 8, 256, 0, stream>>>(chart, accv, len, 0);
    }

    root_init_k<<<1, 256, 0, stream>>>(chart, rootb);

    // ---------------- outside pass ----------------
    for (int len = TT - 1; len >= 1; --len) {
        int S = TT + 1 - len, C = TT - len, rows = S * BB;
        for (int c = 0; c < C; ++c) {
            gemm_k<1, 0><<<dim3(S, 6), 256, 0, stream>>>(chart, nullptr, Wb_bf, nullptr,
                                                         u, nullptr, nullptr, DD, len, c, 1, C);
            score_dot_k<<<rows / 8, 256, 0, stream>>>(chart, u, scrs, len, c, 1, C);
        }
        softmax_k<<<(rows + 255) / 256, 256, 0, stream>>>(chart, scrs, len, 1, S, C);
        zero_k<<<1024, 256, 0, stream>>>(accv, (long long)rows * DD);
        for (int c = 0; c < C; ++c) {
            gemm_k<0, 2><<<dim3(S, 6), 256, 0, stream>>>(chart, nullptr, W1_bf, b1,
                                                         nullptr, h, nullptr, D2, len, c, 1, C);
            gemm_k<2, 3><<<dim3(S, 6), 256, 0, stream>>>(chart, h, W2_bf, b2,
                                                         accv, nullptr, scrs, DD, len, c, 1, C);
        }
        norm_k<<<rows / 8, 256, 0, stream>>>(chart, accv, len, 1);
    }
}
